// GlobalSemanticAwareWeight_81698867904873
// MI455X (gfx1250) — matrix operand order repaired and checked by reading, not compiled, not run on hardware
//
#include <hip/hip_runtime.h>
#include <cmath>

// ---------------------------------------------------------------------------
// CDNA5 (gfx1250) wave32 WMMA types
// ---------------------------------------------------------------------------
typedef __attribute__((ext_vector_type(16))) __bf16 v16bf;
typedef __attribute__((ext_vector_type(8)))  float  v8f;
typedef __attribute__((ext_vector_type(2)))  __bf16 bf16x2;
typedef __attribute__((ext_vector_type(8)))  __bf16 bf16x8;

#define ACT_NONE 0
#define ACT_SILU 1
#define ACT_RELU 2
#define ACT_SOFTPLUS 3

#define LDS_STRIDE 48   // halves; 96B row stride keeps rows 16B aligned

__device__ __forceinline__ float apply_act(float v, int act) {
  switch (act) {
    case ACT_SILU:     return v / (1.f + __expf(-v));
    case ACT_RELU:     return fmaxf(v, 0.f);
    case ACT_SOFTPLUS: return (v > 15.f) ? v : log1pf(__expf(v));
    default:           return v;
  }
}

__device__ __forceinline__ bf16x2 pk2(float a, float b) {
  bf16x2 r;
  r.x = (__bf16)a;
  r.y = (__bf16)b;
  return r;
}

// ---------------------------------------------------------------------------
// Register staging for the GEMM software pipeline.
// A-tile: 128x32 elements.  W-tile: 64x32 fp32 weights.
// ---------------------------------------------------------------------------
template <typename TA> struct StageRegs;
template <> struct StageRegs<float>  { float2 a[8]; float2 w[4]; };
template <> struct StageRegs<__bf16> { bf16x8 a[2]; float2 w[4]; };

// Phase 1: issue all global loads for tile k0 into registers (batched).
template <typename TA>
__device__ __forceinline__ void stage_global(StageRegs<TA>& R, const TA* A, int lda,
                                             const float* W, int rowBase, int colBase,
                                             int M, int N, int K, int k0, int tid);

template <>
__device__ __forceinline__ void stage_global<float>(StageRegs<float>& R, const float* A,
                                                    int lda, const float* W, int rowBase,
                                                    int colBase, int M, int N, int K,
                                                    int k0, int tid) {
  if (k0 + 32 <= K) {
    #pragma unroll
    for (int it = 0; it < 8; ++it) {
      int idx = tid + it * 256, r = idx >> 4, k = (idx & 15) << 1;
      int gr = rowBase + r; gr = (gr < M) ? gr : (M - 1);
      R.a[it] = *(const float2*)&A[(size_t)gr * lda + k0 + k];
    }
    #pragma unroll
    for (int it = 0; it < 4; ++it) {
      int idx = tid + it * 256, r = idx >> 4, k = (idx & 15) << 1;
      int gn = colBase + r; gn = (gn < N) ? gn : (N - 1);
      R.w[it] = *(const float2*)&W[(size_t)gn * K + k0 + k];
    }
    if (k0 + 32 < K) {  // lowers to global_prefetch_b8
      int pr = rowBase + (tid >> 1); pr = (pr < M) ? pr : (M - 1);
      __builtin_prefetch(&A[(size_t)pr * lda + k0 + 32], 0, 0);
    }
  } else {
    // K tail (only the tiny dt-rank GEMMs): clamp address, zero-select.
    #pragma unroll
    for (int it = 0; it < 8; ++it) {
      int idx = tid + it * 256, r = idx >> 4, k = (idx & 15) << 1;
      int gr = rowBase + r; gr = (gr < M) ? gr : (M - 1);
      int k0a = (k0 + k     < K) ? (k0 + k)     : (K - 1);
      int k1a = (k0 + k + 1 < K) ? (k0 + k + 1) : (K - 1);
      float a0 = A[(size_t)gr * lda + k0a];
      float a1 = A[(size_t)gr * lda + k1a];
      if (k0 + k     >= K) a0 = 0.f;
      if (k0 + k + 1 >= K) a1 = 0.f;
      R.a[it].x = a0; R.a[it].y = a1;
    }
    #pragma unroll
    for (int it = 0; it < 4; ++it) {
      int idx = tid + it * 256, r = idx >> 4, k = (idx & 15) << 1;
      int gn = colBase + r; gn = (gn < N) ? gn : (N - 1);
      int k0a = (k0 + k     < K) ? (k0 + k)     : (K - 1);
      int k1a = (k0 + k + 1 < K) ? (k0 + k + 1) : (K - 1);
      float b0 = W[(size_t)gn * K + k0a];
      float b1 = W[(size_t)gn * K + k1a];
      if (k0 + k     >= K) b0 = 0.f;
      if (k0 + k + 1 >= K) b1 = 0.f;
      R.w[it].x = b0; R.w[it].y = b1;
    }
  }
}

template <>
__device__ __forceinline__ void stage_global<__bf16>(StageRegs<__bf16>& R, const __bf16* A,
                                                     int lda, const float* W, int rowBase,
                                                     int colBase, int M, int N, int K,
                                                     int k0, int tid) {
  // All bf16 activation strides are multiples of 32 -> 16B-aligned b128 copies.
  #pragma unroll
  for (int it = 0; it < 2; ++it) {
    int idx = tid + it * 256, r = idx >> 2, kc = (idx & 3) << 3;
    int gr = rowBase + r; gr = (gr < M) ? gr : (M - 1);
    R.a[it] = *(const bf16x8*)&A[(size_t)gr * lda + k0 + kc];
  }
  #pragma unroll
  for (int it = 0; it < 4; ++it) {
    int idx = tid + it * 256, r = idx >> 4, k = (idx & 15) << 1;
    int gn = colBase + r; gn = (gn < N) ? gn : (N - 1);
    R.w[it] = *(const float2*)&W[(size_t)gn * K + k0 + k];
  }
  if (k0 + 32 < K) {
    int pr = rowBase + (tid >> 1); pr = (pr < M) ? pr : (M - 1);
    __builtin_prefetch(&A[(size_t)pr * lda + k0 + 32], 0, 0);
  }
}

// Phase 2: convert (fp32 path) and store registers to LDS.
template <typename TA>
__device__ __forceinline__ void stage_store(const StageRegs<TA>& R,
                                            __bf16 (*sA)[LDS_STRIDE],
                                            __bf16 (*sB)[LDS_STRIDE], int tid);

template <>
__device__ __forceinline__ void stage_store<float>(const StageRegs<float>& R,
                                                   __bf16 (*sA)[LDS_STRIDE],
                                                   __bf16 (*sB)[LDS_STRIDE], int tid) {
  #pragma unroll
  for (int it = 0; it < 8; ++it) {
    int idx = tid + it * 256, r = idx >> 4, k = (idx & 15) << 1;
    *(bf16x2*)&sA[r][k] = pk2(R.a[it].x, R.a[it].y);
  }
  #pragma unroll
  for (int it = 0; it < 4; ++it) {
    int idx = tid + it * 256, r = idx >> 4, k = (idx & 15) << 1;
    *(bf16x2*)&sB[r][k] = pk2(R.w[it].x, R.w[it].y);
  }
}

template <>
__device__ __forceinline__ void stage_store<__bf16>(const StageRegs<__bf16>& R,
                                                    __bf16 (*sA)[LDS_STRIDE],
                                                    __bf16 (*sB)[LDS_STRIDE], int tid) {
  #pragma unroll
  for (int it = 0; it < 2; ++it) {
    int idx = tid + it * 256, r = idx >> 2, kc = (idx & 3) << 3;
    *(bf16x8*)&sA[r][kc] = R.a[it];
  }
  #pragma unroll
  for (int it = 0; it < 4; ++it) {
    int idx = tid + it * 256, r = idx >> 4, k = (idx & 15) << 1;
    *(bf16x2*)&sB[r][k] = pk2(R.w[it].x, R.w[it].y);
  }
}

// ---------------------------------------------------------------------------
// Generic GEMM:  C[M,N] = act(A[M,K] * W[N,K]^T + bias)
// bf16 WMMA 16x16x32, fp32 accumulate. Block = 256 threads (8 waves),
// tile 128(M) x 64(N), K-step 32, per-wave 32x32 (2x2 WMMA tiles).
// Software-pipelined: double LDS buffers, register staging, 1 barrier/step.
// ---------------------------------------------------------------------------
template <typename TA, typename TC>
__global__ __launch_bounds__(256)
void gemm_wmma(const TA* __restrict__ A, int lda,
               const float* __restrict__ W,
               const float* __restrict__ bias,
               TC* __restrict__ Cout, int ldc,
               int M, int N, int K, int act) {
  __shared__ __bf16 sAbuf[2][128][LDS_STRIDE];
  __shared__ __bf16 sBbuf[2][64][LDS_STRIDE];

  const int tid  = threadIdx.x;
  const int lane = tid & 31;
  const int wave = tid >> 5;      // 0..7
  const int wm   = wave >> 1;     // 0..3  : 32-row group
  const int wn   = wave & 1;      // 0..1  : 32-col group
  const int fm   = lane & 15;
  const int hi   = lane >> 4;
  const int khA  = hi * 8;        // A-frag K sub-base (ISA 16-bit A layout)
  const int kbB  = hi * 8;        // B-frag K sub-base (same K order as A)

  const int rowBase = blockIdx.x * 128;
  const int colBase = blockIdx.y * 64;
  const int nk = (K + 31) >> 5;

  v8f acc[2][2] = {};
  StageRegs<TA> R;

  // Prologue: stage tile 0.
  stage_global<TA>(R, A, lda, W, rowBase, colBase, M, N, K, 0, tid);
  stage_store<TA>(R, sAbuf[0], sBbuf[0], tid);

  for (int kt = 0; kt < nk; ++kt) {
    __syncthreads();
    __bf16 (*sA)[LDS_STRIDE] = sAbuf[kt & 1];
    __bf16 (*sB)[LDS_STRIDE] = sBbuf[kt & 1];

    // Fragment loads (ds_load_b128 pairs per the ISA VGPR layouts).
    v16bf afrag[2], bfrag[2];
    #pragma unroll
    for (int tm = 0; tm < 2; ++tm) {
      const int r = wm * 32 + tm * 16 + fm;
      #pragma unroll
      for (int i = 0; i < 8; ++i) {
        int kk = (i < 4) ? (khA + 2 * i) : (16 + khA + 2 * (i - 4));
        afrag[tm][2 * i]     = sA[r][kk];
        afrag[tm][2 * i + 1] = sA[r][kk + 1];
      }
    }
    #pragma unroll
    for (int tn = 0; tn < 2; ++tn) {
      const int ncol = wn * 32 + tn * 16 + fm;
      #pragma unroll
      for (int j = 0; j < 8; ++j) {
        int kb = (j < 4) ? (kbB + 2 * j) : (16 + kbB + 2 * (j - 4));
        bfrag[tn][2 * j]     = sB[ncol][kb];
        bfrag[tn][2 * j + 1] = sB[ncol][kb + 1];
      }
    }

    const bool hasNext = (kt + 1 < nk);
    // Issue next tile's global loads before the math (latency overlap).
    if (hasNext)
      stage_global<TA>(R, A, lda, W, rowBase, colBase, M, N, K, (kt + 1) * 32, tid);

    #pragma unroll
    for (int tm = 0; tm < 2; ++tm)
      #pragma unroll
      for (int tn = 0; tn < 2; ++tn)
        acc[tm][tn] = __builtin_amdgcn_wmma_f32_16x16x32_bf16(
            false, afrag[tm], false, bfrag[tn],
            (short)0, acc[tm][tn], false, false);

    // Store next tile into the other LDS buffer (visible after next barrier).
    if (hasNext)
      stage_store<TA>(R, sAbuf[(kt + 1) & 1], sBbuf[(kt + 1) & 1], tid);
  }

  // C/D layout: VGPR r, lanes0-15 -> M=r, lanes16-31 -> M=8+r; N = lane&15.
  const int cn = lane & 15;
  const int mh = hi * 8;
  #pragma unroll
  for (int tm = 0; tm < 2; ++tm)
    #pragma unroll
    for (int tn = 0; tn < 2; ++tn)
      #pragma unroll
      for (int r = 0; r < 8; ++r) {
        int row = rowBase + wm * 32 + tm * 16 + mh + r;
        int col = colBase + wn * 32 + tn * 16 + cn;
        if (row < M && col < N) {
          float v = acc[tm][tn][r];
          if (bias) v += bias[col];
          Cout[(size_t)row * ldc + col] = (TC)apply_act(v, act);
        }
      }
}

// ---------------------------------------------------------------------------
// Depthwise causal conv (D_CONV=4) + bias + SiLU:  xz[:,0:Din] -> xc
// ---------------------------------------------------------------------------
__global__ __launch_bounds__(256)
void conv_silu(const __bf16* __restrict__ xz, int ldxz,
               const float* __restrict__ cw, const float* __restrict__ cb,
               __bf16* __restrict__ xc, int Din, int L) {
  size_t idx = (size_t)blockIdx.x * blockDim.x + threadIdx.x;
  if (idx >= (size_t)L * Din) return;
  int c = (int)(idx % Din);
  int t = (int)(idx / Din);
  float acc = cb[c];
  #pragma unroll
  for (int k = 0; k < 4; ++k) {
    int tt = t - 3 + k;
    if (tt >= 0) acc += (float)xz[(size_t)tt * ldxz + c] * cw[c * 4 + k];
  }
  xc[idx] = (__bf16)(acc / (1.f + __expf(-acc)));
}

// ---------------------------------------------------------------------------
// Chunked selective-scan, pass 1: local scan per chunk.
// Lane layout: s = tid&15 (state), c = blockIdx.y*16 + tid>>4 (channel).
// ---------------------------------------------------------------------------
__global__ __launch_bounds__(256)
void scan_pass1(const __bf16* __restrict__ xc, const float* __restrict__ dt,
                const float* __restrict__ dbl, int ldb, int dtr,
                const float* __restrict__ A_log, int Din, int L, int chunkLen,
                float* __restrict__ ys, float* __restrict__ chunkA,
                float* __restrict__ chunkH) {
  const int s = threadIdx.x & 15;
  const int c = blockIdx.y * 16 + (threadIdx.x >> 4);
  const int chunk = blockIdx.x;
  const int t0 = chunk * chunkLen;
  const int t1 = (t0 + chunkLen < L) ? (t0 + chunkLen) : L;
  const float Acs = -__expf(A_log[c * 16 + s]);
  float h = 0.f, p = 1.f;
  for (int t = t0; t < t1; ++t) {
    float dtv = dt[(size_t)t * Din + c];
    float a   = __expf(dtv * Acs);
    float bx  = dtv * (float)xc[(size_t)t * Din + c] *
                dbl[(size_t)t * ldb + dtr + s];
    h = a * h + bx;
    p *= a;
    float contrib = h * dbl[(size_t)t * ldb + dtr + 16 + s];
    #pragma unroll
    for (int off = 8; off; off >>= 1) contrib += __shfl_xor(contrib, off, 16);
    if (s == 0) ys[(size_t)t * Din + c] = contrib;
  }
  size_t idx = (size_t)chunk * Din * 16 + (size_t)c * 16 + s;
  chunkA[idx] = p;
  chunkH[idx] = h;
}

// Pass 2: tiny sequential carry across chunks (one thread per (c,s) stream).
__global__ __launch_bounds__(256)
void scan_carry(const float* __restrict__ chunkA, const float* __restrict__ chunkH,
                float* __restrict__ carryIn, int numChunks, int total) {
  int i = blockIdx.x * blockDim.x + threadIdx.x;
  if (i >= total) return;
  float hc = 0.f;
  for (int k = 0; k < numChunks; ++k) {
    size_t idx = (size_t)k * total + i;
    carryIn[idx] = hc;
    hc = chunkA[idx] * hc + chunkH[idx];
  }
}

// Pass 3: correction  y_t += (prefix-prod a) * H_in dotted with C_t.
__global__ __launch_bounds__(256)
void scan_pass3(const float* __restrict__ dt, const float* __restrict__ dbl,
                int ldb, int dtr, const float* __restrict__ A_log,
                const float* __restrict__ carryIn, int Din, int L, int chunkLen,
                float* __restrict__ ys) {
  const int s = threadIdx.x & 15;
  const int c = blockIdx.y * 16 + (threadIdx.x >> 4);
  const int chunk = blockIdx.x + 1;   // chunk 0 has zero carry
  const int t0 = chunk * chunkLen;
  const int t1 = (t0 + chunkLen < L) ? (t0 + chunkLen) : L;
  const float Acs = -__expf(A_log[c * 16 + s]);
  const float Hin = carryIn[(size_t)chunk * Din * 16 + (size_t)c * 16 + s];
  float p = 1.f;
  for (int t = t0; t < t1; ++t) {
    p *= __expf(dt[(size_t)t * Din + c] * Acs);
    float contrib = p * Hin * dbl[(size_t)t * ldb + dtr + 16 + s];
    #pragma unroll
    for (int off = 8; off; off >>= 1) contrib += __shfl_xor(contrib, off, 16);
    if (s == 0) ys[(size_t)t * Din + c] += contrib;
  }
}

// y = (ys + xc * D) * silu(z),  z = xz[:, Din:2*Din]
__global__ __launch_bounds__(256)
void combine_gate(const float* __restrict__ ys, const __bf16* __restrict__ xc,
                  const __bf16* __restrict__ xz, int ldxz,
                  const float* __restrict__ Dv, __bf16* __restrict__ y,
                  int Din, size_t n) {
  size_t idx = (size_t)blockIdx.x * blockDim.x + threadIdx.x;
  if (idx >= n) return;
  int c = (int)(idx % Din);
  size_t t = idx / Din;
  float z = (float)xz[t * (size_t)ldxz + Din + c];
  float v = (ys[idx] + (float)xc[idx] * Dv[c]) * (z / (1.f + __expf(-z)));
  y[idx] = (__bf16)v;
}

// Final head:  out[t] = sigmoid(h[t,:] . w + b).  One wave per row.
__global__ __launch_bounds__(256)
void head_sigmoid(const __bf16* __restrict__ h, const float* __restrict__ w,
                  const float* __restrict__ b, float* __restrict__ out, int M) {
  int wave = threadIdx.x >> 5;
  int lane = threadIdx.x & 31;
  int row  = blockIdx.x * 8 + wave;
  if (row >= M) return;
  float acc = 0.f;
  #pragma unroll
  for (int i = lane; i < 128; i += 32) acc += (float)h[(size_t)row * 128 + i] * w[i];
  #pragma unroll
  for (int off = 16; off; off >>= 1) acc += __shfl_xor(acc, off, 32);
  if (lane == 0) out[row] = 1.f / (1.f + __expf(-(acc + b[0])));
}

// ---------------------------------------------------------------------------
// Host side
// ---------------------------------------------------------------------------
static inline void* bumpB(char* base, size_t& off, size_t bytes) {
  void* p = (void*)(base + off);
  off += ((bytes + 255) & ~(size_t)255);
  return p;
}
static inline float* bumpF(char* base, size_t& off, size_t elems) {
  return (float*)bumpB(base, off, elems * sizeof(float));
}
static inline __bf16* bumpH(char* base, size_t& off, size_t elems) {
  return (__bf16*)bumpB(base, off, elems * sizeof(__bf16));
}

template <typename TA, typename TC>
static void launch_gemm(hipStream_t st, const TA* A, int lda, const float* W,
                        const float* bias, TC* C, int ldc,
                        int M, int N, int K, int act) {
  dim3 grid((unsigned)((M + 127) / 128), (unsigned)((N + 63) / 64));
  gemm_wmma<TA, TC><<<grid, dim3(256), 0, st>>>(A, lda, W, bias, C, ldc, M, N, K, act);
}

struct MambaP {
  const float *in_proj, *conv_w, *conv_b, *x_proj, *dt_w, *dt_b, *A_log, *D, *out_proj;
};

template <typename TSrc>
static void run_mamba_level(hipStream_t st, int V, int Dm, int Din, int dtr,
                            const TSrc* src, int srcC,
                            const float* pw, const float* pb,
                            const MambaP& mp,
                            __bf16* fused, int fusedLd, int fusedCol,
                            char* base, size_t scratchOff) {
  size_t o = scratchOff;
  const int ldb = dtr + 32;
  const int CL = 2048;
  const int NC = (V + CL - 1) / CL;
  __bf16* xin = bumpH(base, o, (size_t)V * Dm);
  __bf16* xz  = bumpH(base, o, (size_t)V * 2 * Din);
  __bf16* xc  = bumpH(base, o, (size_t)V * Din);
  float*  dbl = bumpF(base, o, (size_t)V * ldb);
  float*  dtb = bumpF(base, o, (size_t)V * Din);
  float*  ys  = bumpF(base, o, (size_t)V * Din);
  __bf16* yb  = bumpH(base, o, (size_t)V * Din);
  float*  cA  = bumpF(base, o, (size_t)NC * Din * 16);
  float*  cH  = bumpF(base, o, (size_t)NC * Din * 16);
  float*  cI  = bumpF(base, o, (size_t)NC * Din * 16);

  // projection into mamba d_model
  launch_gemm(st, src, srcC, pw, pb, xin, Dm, V, Dm, srcC, ACT_NONE);
  // in_proj -> [x | z]
  launch_gemm(st, xin, Dm, mp.in_proj, (const float*)nullptr, xz, 2 * Din,
              V, 2 * Din, Dm, ACT_NONE);
  // depthwise conv + SiLU
  {
    size_t n = (size_t)V * Din;
    conv_silu<<<dim3((unsigned)((n + 255) / 256)), dim3(256), 0, st>>>(
        xz, 2 * Din, mp.conv_w, mp.conv_b, xc, Din, V);
  }
  // x_proj -> [dt_low | B | C]   (fp32 out: feeds scan + dt GEMM)
  launch_gemm(st, xc, Din, mp.x_proj, (const float*)nullptr, dbl, ldb,
              V, ldb, Din, ACT_NONE);
  // dt = softplus(dt_low @ dt_w^T + dt_b)
  launch_gemm(st, dbl, ldb, mp.dt_w, mp.dt_b, dtb, Din, V, Din, dtr, ACT_SOFTPLUS);
  // chunked scan
  scan_pass1<<<dim3((unsigned)NC, (unsigned)(Din / 16)), dim3(256), 0, st>>>(
      xc, dtb, dbl, ldb, dtr, mp.A_log, Din, V, CL, ys, cA, cH);
  {
    int total = Din * 16;
    scan_carry<<<dim3((unsigned)((total + 255) / 256)), dim3(256), 0, st>>>(
        cA, cH, cI, NC, total);
  }
  if (NC > 1)
    scan_pass3<<<dim3((unsigned)(NC - 1), (unsigned)(Din / 16)), dim3(256), 0, st>>>(
        dtb, dbl, ldb, dtr, mp.A_log, cI, Din, V, CL, ys);
  // gate + skip
  {
    size_t n = (size_t)V * Din;
    combine_gate<<<dim3((unsigned)((n + 255) / 256)), dim3(256), 0, st>>>(
        ys, xc, xz, 2 * Din, mp.D, yb, Din, n);
  }
  // out_proj straight into the fused buffer column slice
  launch_gemm(st, yb, Din, mp.out_proj, (const float*)nullptr, fused + fusedCol,
              fusedLd, V, Dm, Din, ACT_NONE);
}

extern "C" void kernel_launch(void* const* d_in, const int* in_sizes, int n_in,
                              void* d_out, int out_size, void* d_ws, size_t ws_size,
                              hipStream_t stream) {
  const float* f1     = (const float*)d_in[0];
  const float* ds2_w  = (const float*)d_in[1];
  const float* ds2_b  = (const float*)d_in[2];
  const float* ds4_w  = (const float*)d_in[3];
  const float* ds4_b  = (const float*)d_in[4];
  const float* p1_w   = (const float*)d_in[5];
  const float* p1_b   = (const float*)d_in[6];
  const float* p2_w   = (const float*)d_in[7];
  const float* p2_b   = (const float*)d_in[8];
  const float* p4_w   = (const float*)d_in[9];
  const float* p4_b   = (const float*)d_in[10];
  MambaP m1 = {(const float*)d_in[11], (const float*)d_in[12], (const float*)d_in[13],
               (const float*)d_in[14], (const float*)d_in[15], (const float*)d_in[16],
               (const float*)d_in[17], (const float*)d_in[18], (const float*)d_in[19]};
  MambaP m2 = {(const float*)d_in[20], (const float*)d_in[21], (const float*)d_in[22],
               (const float*)d_in[23], (const float*)d_in[24], (const float*)d_in[25],
               (const float*)d_in[26], (const float*)d_in[27], (const float*)d_in[28]};
  MambaP m4 = {(const float*)d_in[29], (const float*)d_in[30], (const float*)d_in[31],
               (const float*)d_in[32], (const float*)d_in[33], (const float*)d_in[34],
               (const float*)d_in[35], (const float*)d_in[36], (const float*)d_in[37]};
  const float* h1_w = (const float*)d_in[38];
  const float* h1_b = (const float*)d_in[39];
  const float* h2_w = (const float*)d_in[40];
  const float* h2_b = (const float*)d_in[41];

  const int V = in_sizes[0] / 256;     // 100000
  char* base = (char*)d_ws;
  size_t off = 0;

  __bf16* f2    = bumpH(base, off, (size_t)V * 128);
  __bf16* f4    = bumpH(base, off, (size_t)V * 64);
  __bf16* fused = bumpH(base, off, (size_t)V * 224);
  __bf16* hbuf  = bumpH(base, off, (size_t)V * 128);
  const size_t scratch0 = off;   // per-level scratch region, reused

  // Downsample chain
  launch_gemm(stream, f1, 256, ds2_w, ds2_b, f2, 128, V, 128, 256, ACT_NONE);
  launch_gemm(stream, f2, 128, ds4_w, ds4_b, f4, 64, V, 64, 128, ACT_NONE);

  // Three Mamba levels (sequential; scratch reused)
  run_mamba_level(stream, V, 128, 256, 8, f1, 256, p1_w, p1_b, m1, fused, 224, 0,   base, scratch0);
  run_mamba_level(stream, V, 64,  128, 4, f2, 128, p2_w, p2_b, m2, fused, 224, 128, base, scratch0);
  run_mamba_level(stream, V, 32,  64,  2, f4, 64,  p4_w, p4_b, m4, fused, 224, 192, base, scratch0);

  // Fusion MLP: h = relu(fused @ h1_w^T + h1_b)
  launch_gemm(stream, fused, 224, h1_w, h1_b, hbuf, 128, V, 128, 224, ACT_RELU);

  // Head: sigmoid(h @ h2_w^T + h2_b)
  head_sigmoid<<<dim3((unsigned)((V + 7) / 8)), dim3(256), 0, stream>>>(
      hbuf, h2_w, h2_b, (float*)d_out, V);
}
